// EncoderLayer_72859825209412
// MI455X (gfx1250) — matrix operand order repaired and checked by reading, not compiled, not run on hardware
//
#include <hip/hip_runtime.h>

#define S_DIM 4096
#define D_DIM 512
#define F_DIM 2048
#define B_DIM 8

typedef __attribute__((ext_vector_type(16))) __bf16 v16bf;
typedef __attribute__((ext_vector_type(8)))  float  v8f;

union Frag {
  v16bf v;
  unsigned int u[8];
};

__device__ __forceinline__ unsigned short f2bf(float f) {
  unsigned int u = __float_as_uint(f);
  u = (u + 0x7FFFu + ((u >> 16) & 1u)) >> 16;
  return (unsigned short)u;
}

__device__ __forceinline__ v8f wmma_bf16(const Frag& a, const Frag& b, v8f c) {
  return __builtin_amdgcn_wmma_f32_16x16x32_bf16(false, a.v, false, b.v,
                                                 (short)0, c, false, false);
}

// A-matrix 16x32 bf16 per-lane K base for VGPR v, lane-half hi (contiguous: k = 16*hi + 2*v)
__device__ __forceinline__ int a_kbase(int v, int hi) {
  return 2 * v + hi * 16;
}

// CDNA5 async copy: 16 bytes global -> LDS, tracked by ASYNCcnt.
// lds_off = wave-relative LDS byte address (low 32 bits of generic shared ptr).
__device__ __forceinline__ void async_ld_b128(unsigned int lds_off, const void* g) {
  asm volatile("global_load_async_to_lds_b128 %0, %1, off"
               :: "v"(lds_off), "v"((unsigned long long)(uintptr_t)g)
               : "memory");
}
__device__ __forceinline__ void wait_async0() {
  asm volatile("s_wait_asynccnt 0x0" ::: "memory");
}

// ---------------------------------------------------------------------------
// Generic bf16 GEMM: C[M,N] = A[M,K] * B[K,N], fp32 accum, epilogue variants.
// Double-buffered LDS; A tile moved with global_load_async_to_lds_b128.
// epi 0: bf16 row-major        epi 1: bf16 relu(acc+bias)
// epi 2: bf16 transposed per batch: out[(b*D + n)*S + s]
// epi 3: f32 out[(b*D+n)*S+s] = res[idx] + acc
// epi 4: f32 out[(b*D+n)*S+s] = res[idx] + acc + bias[n]
// ---------------------------------------------------------------------------
__global__ __launch_bounds__(256) void gemm_bf16_kernel(
    const unsigned short* __restrict__ A, const unsigned short* __restrict__ Bw,
    int M, int N, int K, int epi, void* __restrict__ outp,
    const float* __restrict__ bias, const float* __restrict__ res)
{
  __shared__ unsigned short sA[2][128 * 32];
  __shared__ unsigned short sBt[2][128 * 32];   // transposed: [n][k]

  const int tid  = threadIdx.x;
  const int lane = tid & 31;
  const int wave = tid >> 5;
  const int hi   = lane >> 4;
  const int ln   = lane & 15;
  const int waveM = wave >> 1;   // 0..3
  const int waveN = wave & 1;    // 0..1

  const int rowBlk = blockIdx.y * 128;
  const int colBlk = blockIdx.x * 128;

  v8f acc[2][4];
  for (int i = 0; i < 2; ++i)
    for (int j = 0; j < 4; ++j)
      for (int e = 0; e < 8; ++e) acc[i][j][e] = 0.0f;

  const int arow   = tid >> 1;          // 0..127
  const int ahalf  = (tid & 1) * 16;    // 0/16
  const int bkrow  = tid >> 3;          // 0..31
  const int bchunk = (tid & 7) * 16;    // 0..112

  // --- prologue: stage tile 0 ---
  {
    unsigned int lds = (unsigned int)(uintptr_t)&sA[0][arow * 32 + ahalf];
    const unsigned short* g = A + (size_t)(rowBlk + arow) * K + ahalf;
    async_ld_b128(lds, g);
    async_ld_b128(lds + 16, g + 8);

    const uint4* src = (const uint4*)(Bw + (size_t)bkrow * N + colBlk + bchunk);
    unsigned short tmp[16];
    *(uint4*)&tmp[0] = src[0];
    *(uint4*)&tmp[8] = src[1];
    #pragma unroll
    for (int j = 0; j < 16; ++j)
      sBt[0][(bchunk + j) * 32 + bkrow] = tmp[j];
  }
  wait_async0();
  __syncthreads();

  int buf = 0;
  for (int kk = 0; kk < K; kk += 32) {
    const int nxt = buf ^ 1;
    const bool more = (kk + 32 < K);   // uniform

    uint4 bd0, bd1;
    if (more) {
      // prefetch next A tile straight into LDS (async), next B tile into regs
      unsigned int lds = (unsigned int)(uintptr_t)&sA[nxt][arow * 32 + ahalf];
      const unsigned short* g = A + (size_t)(rowBlk + arow) * K + (kk + 32) + ahalf;
      async_ld_b128(lds, g);
      async_ld_b128(lds + 16, g + 8);
      const uint4* src = (const uint4*)(Bw + (size_t)(kk + 32 + bkrow) * N + colBlk + bchunk);
      bd0 = src[0]; bd1 = src[1];
    }

    // --- compute on current buffer ---
    Frag af[2];
    #pragma unroll
    for (int mt = 0; mt < 2; ++mt) {
      const int row = waveM * 32 + mt * 16 + ln;
      #pragma unroll
      for (int v = 0; v < 8; ++v)
        af[mt].u[v] = *(const unsigned int*)&sA[buf][row * 32 + a_kbase(v, hi)];
    }
    #pragma unroll
    for (int nt = 0; nt < 4; ++nt) {
      Frag bfr;
      const int col = waveN * 64 + nt * 16 + ln;
      #pragma unroll
      for (int v = 0; v < 8; ++v)
        bfr.u[v] = *(const unsigned int*)&sBt[buf][col * 32 + hi * 16 + 2 * v];
      acc[0][nt] = wmma_bf16(af[0], bfr, acc[0][nt]);
      acc[1][nt] = wmma_bf16(af[1], bfr, acc[1][nt]);
    }

    if (more) {
      // scatter next B tile (transposed) into the other buffer
      unsigned short tmp[16];
      *(uint4*)&tmp[0] = bd0;
      *(uint4*)&tmp[8] = bd1;
      #pragma unroll
      for (int j = 0; j < 16; ++j)
        sBt[nxt][(bchunk + j) * 32 + bkrow] = tmp[j];
    }

    wait_async0();
    __syncthreads();
    buf = nxt;
  }

  #pragma unroll
  for (int mt = 0; mt < 2; ++mt) {
    #pragma unroll
    for (int nt = 0; nt < 4; ++nt) {
      #pragma unroll
      for (int j = 0; j < 8; ++j) {
        const int gr = rowBlk + waveM * 32 + mt * 16 + j + 8 * hi;
        const int gc = colBlk + waveN * 64 + nt * 16 + ln;
        const float v = acc[mt][nt][j];
        if (epi == 0) {
          ((unsigned short*)outp)[(size_t)gr * N + gc] = f2bf(v);
        } else if (epi == 1) {
          float t = v + bias[gc];
          ((unsigned short*)outp)[(size_t)gr * N + gc] = f2bf(t > 0.0f ? t : 0.0f);
        } else if (epi == 2) {
          const int b = gr >> 12, s = gr & 4095;
          ((unsigned short*)outp)[((size_t)(b * D_DIM + gc)) * S_DIM + s] = f2bf(v);
        } else if (epi == 3) {
          const int b = gr >> 12, s = gr & 4095;
          const size_t idx = ((size_t)(b * D_DIM + gc)) * S_DIM + s;
          ((float*)outp)[idx] = res[idx] + v;
        } else {
          const int b = gr >> 12, s = gr & 4095;
          const size_t idx = ((size_t)(b * D_DIM + gc)) * S_DIM + s;
          ((float*)outp)[idx] = res[idx] + v + bias[gc];
        }
      }
    }
  }
}

// ---------------------------------------------------------------------------
// Fused flash attention: Ctx[b,s,:] = softmax(Q Kt / sqrt(D)) V
// One workgroup = 16 query rows; 8 waves; online softmax over 128-key blocks.
// Q,K: [B*S, D] bf16 row-major.  Vt: [B, D, S] bf16.  Ctx: [B*S, D] bf16.
// ---------------------------------------------------------------------------
__global__ __launch_bounds__(256) void attention_kernel(
    const unsigned short* __restrict__ Q,
    const unsigned short* __restrict__ Km,
    const unsigned short* __restrict__ Vt,
    unsigned short* __restrict__ Ctx)
{
  __shared__ unsigned short sQ[16 * 512];
  __shared__ unsigned short sP[16 * 128];
  __shared__ float sWmax[8][16];
  __shared__ float sWsum[8][16];
  __shared__ float sM[16];
  __shared__ float sL[16];
  __shared__ float sScale[16];

  const int tid  = threadIdx.x;
  const int lane = tid & 31;
  const int wave = tid >> 5;
  const int hi   = lane >> 4;
  const int ln   = lane & 15;
  const int b    = blockIdx.y;
  const int q0   = blockIdx.x * 16;

  { // async-stage 16x512 Q tile into LDS
    const int row = tid >> 4;
    const int chunk = (tid & 15) * 32;
    unsigned int lds = (unsigned int)(uintptr_t)&sQ[row * 512 + chunk];
    const unsigned short* g = Q + ((size_t)(b * S_DIM + q0 + row)) * D_DIM + chunk;
    #pragma unroll
    for (int i = 0; i < 4; ++i)
      async_ld_b128(lds + 16 * i, g + 8 * i);
  }
  if (tid < 16) { sM[tid] = -3.0e38f; sL[tid] = 0.0f; }

  v8f acc[4];
  for (int nt = 0; nt < 4; ++nt)
    for (int e = 0; e < 8; ++e) acc[nt][e] = 0.0f;

  wait_async0();
  __syncthreads();

  const float scl = 0.044194173824159216f;   // 1/sqrt(512)

  for (int kb = 0; kb < S_DIM; kb += 128) {
    // phase 1: per-wave 16x16 raw score tile (keys kb + wave*16 ..)
    v8f sc;
    for (int e = 0; e < 8; ++e) sc[e] = 0.0f;
    const size_t krow = (size_t)(b * S_DIM + kb + wave * 16 + ln) * D_DIM;
    if (kb + 128 < S_DIM)   // warm GL2 for the next key block
      __builtin_prefetch(&Km[krow + (size_t)128 * D_DIM], 0, 1);
    for (int kk = 0; kk < D_DIM; kk += 32) {
      Frag aq, bk;
      #pragma unroll
      for (int v = 0; v < 8; ++v)
        aq.u[v] = *(const unsigned int*)&sQ[ln * 512 + kk + a_kbase(v, hi)];
      #pragma unroll
      for (int v = 0; v < 8; ++v)
        bk.u[v] = *(const unsigned int*)&Km[krow + kk + hi * 16 + 2 * v];
      sc = wmma_bf16(aq, bk, sc);
    }
    float mx[8];
    #pragma unroll
    for (int j = 0; j < 8; ++j) {
      float t = sc[j] * scl;
      sc[j] = t;
      #pragma unroll
      for (int off = 1; off < 16; off <<= 1)
        t = fmaxf(t, __shfl_xor(t, off, 32));
      mx[j] = t;
    }
    if (ln == 0) {
      #pragma unroll
      for (int j = 0; j < 8; ++j) sWmax[wave][j + 8 * hi] = mx[j];
    }
    __syncthreads();

    // phase 2: new running max + rescale factor
    if (tid < 16) {
      float m = sM[tid];
      float bm = sWmax[0][tid];
      #pragma unroll
      for (int w = 1; w < 8; ++w) bm = fmaxf(bm, sWmax[w][tid]);
      const float mn = fmaxf(m, bm);
      sScale[tid] = __expf(m - mn);
      sM[tid] = mn;
    }
    __syncthreads();

    // phase 3: p = exp(s - m), row sums, stage P (bf16), rescale accumulators
    #pragma unroll
    for (int j = 0; j < 8; ++j) {
      const int row = j + 8 * hi;
      const float p = __expf(sc[j] - sM[row]);
      sP[row * 128 + wave * 16 + ln] = f2bf(p);
      float t = p;
      #pragma unroll
      for (int off = 1; off < 16; off <<= 1)
        t += __shfl_xor(t, off, 32);
      if (ln == 0) sWsum[wave][row] = t;
    }
    #pragma unroll
    for (int j = 0; j < 8; ++j) {
      const float scj = sScale[j + 8 * hi];
      #pragma unroll
      for (int nt = 0; nt < 4; ++nt) acc[nt][j] *= scj;
    }
    __syncthreads();

    // phase 4: running denominator
    if (tid < 16) {
      float l = sL[tid] * sScale[tid];
      #pragma unroll
      for (int w = 0; w < 8; ++w) l += sWsum[w][tid];
      sL[tid] = l;
    }

    // phase 5: P @ V, each wave owns 64 output columns
    #pragma unroll
    for (int kt = 0; kt < 4; ++kt) {
      Frag ap;
      #pragma unroll
      for (int v = 0; v < 8; ++v)
        ap.u[v] = *(const unsigned int*)&sP[ln * 128 + kt * 32 + a_kbase(v, hi)];
      #pragma unroll
      for (int nt = 0; nt < 4; ++nt) {
        Frag bv;
        const int d = wave * 64 + nt * 16 + ln;
        const size_t vb = ((size_t)(b * D_DIM + d)) * S_DIM + kb + kt * 32 + hi * 16;
        #pragma unroll
        for (int v = 0; v < 8; ++v)
          bv.u[v] = *(const unsigned int*)&Vt[vb + 2 * v];
        acc[nt] = wmma_bf16(ap, bv, acc[nt]);
      }
    }
    __syncthreads();
  }

  #pragma unroll
  for (int nt = 0; nt < 4; ++nt) {
    #pragma unroll
    for (int j = 0; j < 8; ++j) {
      const int row = j + 8 * hi;
      const float val = acc[nt][j] / sL[row];
      Ctx[((size_t)(b * S_DIM + q0 + row)) * D_DIM + wave * 64 + nt * 16 + ln] = f2bf(val);
    }
  }
}

// ---------------------------------------------------------------------------
// Per-batch sum / sumsq reduction (for global mean + L2 norm)
// ---------------------------------------------------------------------------
__global__ __launch_bounds__(256) void reduce_stats_kernel(
    const float* __restrict__ y, double* __restrict__ stats)
{
  __shared__ double ssum[256], ssq[256];
  const int b = blockIdx.y;
  const float* p = y + (size_t)b * (D_DIM * S_DIM);
  double s = 0.0, q = 0.0;
  for (int i = blockIdx.x * 256 + threadIdx.x; i < D_DIM * S_DIM; i += gridDim.x * 256) {
    const float v = p[i];
    s += v; q += (double)v * v;
  }
  ssum[threadIdx.x] = s; ssq[threadIdx.x] = q;
  __syncthreads();
  for (int off = 128; off > 0; off >>= 1) {
    if (threadIdx.x < off) {
      ssum[threadIdx.x] += ssum[threadIdx.x + off];
      ssq[threadIdx.x]  += ssq[threadIdx.x + off];
    }
    __syncthreads();
  }
  if (threadIdx.x == 0) {
    atomicAdd(&stats[b * 2 + 0], ssum[0]);
    atomicAdd(&stats[b * 2 + 1], ssq[0]);
  }
}

__global__ void zero_stats_kernel(double* stats) {
  if (threadIdx.x < 2 * B_DIM) stats[threadIdx.x] = 0.0;
}

// normalize in place; optionally emit transposed bf16 copy [B,S,D]
__global__ void normalize_kernel(float* __restrict__ y,
                                 const double* __restrict__ stats,
                                 unsigned short* __restrict__ post_t)
{
  const size_t idx = (size_t)blockIdx.x * 256 + threadIdx.x;
  const size_t per = (size_t)D_DIM * S_DIM;           // 2^21
  const int b = (int)(idx >> 21);
  const size_t rem = idx & (per - 1);
  const double mean = stats[b * 2] / (double)per;
  double l2d = stats[b * 2 + 1] - mean * mean * (double)per;
  const float l2 = sqrtf(l2d > 0.0 ? (float)l2d : 0.0f);
  const float val = (y[idx] - (float)mean) / (l2 + 1e-7f);
  y[idx] = val;
  if (post_t) {
    const int d = (int)(rem >> 12);
    const int s = (int)(rem & 4095);
    post_t[((size_t)(b * S_DIM + s)) * D_DIM + d] = f2bf(val);
  }
}

__global__ void cast_f32_bf16_kernel(const float* __restrict__ src,
                                     unsigned short* __restrict__ dst, int n)
{
  const int i = blockIdx.x * 256 + threadIdx.x;
  if (i < n) dst[i] = f2bf(src[i]);
}

__global__ void transpose_cast_x_kernel(const float* __restrict__ x,
                                        unsigned short* __restrict__ xt)
{
  const size_t idx = (size_t)blockIdx.x * 256 + threadIdx.x;
  const int b = (int)(idx >> 21);
  const size_t rem = idx & ((1u << 21) - 1);
  const int d = (int)(rem >> 12);
  const int s = (int)(rem & 4095);
  xt[((size_t)(b * S_DIM + s)) * D_DIM + d] = f2bf(x[idx]);
}

// ---------------------------------------------------------------------------
extern "C" void kernel_launch(void* const* d_in, const int* in_sizes, int n_in,
                              void* d_out, int out_size, void* d_ws, size_t ws_size,
                              hipStream_t stream)
{
  (void)in_sizes; (void)n_in; (void)out_size; (void)ws_size;

  const float* x  = (const float*)d_in[0];
  const float* b1 = (const float*)d_in[6];
  const float* b2 = (const float*)d_in[8];

  char* ws = (char*)d_ws;
  const size_t MB = 1ull << 20;

  unsigned short* wq = (unsigned short*)(ws + 0);
  unsigned short* wk = (unsigned short*)(ws + 524288);
  unsigned short* wv = (unsigned short*)(ws + 1048576);
  unsigned short* wo = (unsigned short*)(ws + 1572864);
  unsigned short* w1 = (unsigned short*)(ws + 2 * MB);
  unsigned short* w2 = (unsigned short*)(ws + 4 * MB);
  unsigned short* xt  = (unsigned short*)(ws + 6 * MB);    // [B*S, D] bf16; reused as post_t
  unsigned short* qb  = (unsigned short*)(ws + 38 * MB);   // [B*S, D]
  unsigned short* kbf = (unsigned short*)(ws + 70 * MB);   // [B*S, D]
  unsigned short* vtb = (unsigned short*)(ws + 102 * MB);  // [B, D, S]
  unsigned short* ctx = (unsigned short*)(ws + 134 * MB);  // [B*S, D]
  float*          yb  = (float*)(ws + 166 * MB);           // [B, D, S] f32
  unsigned short* h1  = (unsigned short*)(ws + 38 * MB);   // [B*S, F] (reuses q/k/v/ctx)
  double*         st  = (double*)(ws + 230 * MB);          // [B][2]

  const int M = B_DIM * S_DIM;   // 32768
  const int DS = 1 << 21;        // D*S elems per batch

  // 1) casts
  cast_f32_bf16_kernel<<<1024, 256, 0, stream>>>((const float*)d_in[1], wq, D_DIM * D_DIM);
  cast_f32_bf16_kernel<<<1024, 256, 0, stream>>>((const float*)d_in[2], wk, D_DIM * D_DIM);
  cast_f32_bf16_kernel<<<1024, 256, 0, stream>>>((const float*)d_in[3], wv, D_DIM * D_DIM);
  cast_f32_bf16_kernel<<<1024, 256, 0, stream>>>((const float*)d_in[4], wo, D_DIM * D_DIM);
  cast_f32_bf16_kernel<<<4096, 256, 0, stream>>>((const float*)d_in[5], w1, D_DIM * F_DIM);
  cast_f32_bf16_kernel<<<4096, 256, 0, stream>>>((const float*)d_in[7], w2, F_DIM * D_DIM);
  transpose_cast_x_kernel<<<(B_DIM * DS) / 256, 256, 0, stream>>>(x, xt);

  // 2) Q, K, V projections
  gemm_bf16_kernel<<<dim3(D_DIM / 128, M / 128), 256, 0, stream>>>(
      xt, wq, M, D_DIM, D_DIM, 0, qb, nullptr, nullptr);
  gemm_bf16_kernel<<<dim3(D_DIM / 128, M / 128), 256, 0, stream>>>(
      xt, wk, M, D_DIM, D_DIM, 0, kbf, nullptr, nullptr);
  gemm_bf16_kernel<<<dim3(D_DIM / 128, M / 128), 256, 0, stream>>>(
      xt, wv, M, D_DIM, D_DIM, 2, vtb, nullptr, nullptr);

  // 3) fused flash attention -> ctx
  attention_kernel<<<dim3(S_DIM / 16, B_DIM), 256, 0, stream>>>(qb, kbf, vtb, ctx);

  // 4) y = x + (ctx @ Wo)^T  (f32, [B,D,S])
  gemm_bf16_kernel<<<dim3(D_DIM / 128, M / 128), 256, 0, stream>>>(
      ctx, wo, M, D_DIM, D_DIM, 3, yb, nullptr, x);

  // 5) post = norm(y), also post_t bf16 (overwrites xt)
  zero_stats_kernel<<<1, 64, 0, stream>>>(st);
  reduce_stats_kernel<<<dim3(64, B_DIM), 256, 0, stream>>>(yb, st);
  normalize_kernel<<<(B_DIM * DS) / 256, 256, 0, stream>>>(yb, st, xt);

  // 6) FFN: h1 = relu(post_t @ W1 + b1)
  gemm_bf16_kernel<<<dim3(F_DIM / 128, M / 128), 256, 0, stream>>>(
      xt, w1, M, F_DIM, D_DIM, 1, h1, b1, nullptr);

  // 7) z = post + (h1 @ W2 + b2)^T  -> d_out (pre-norm)
  gemm_bf16_kernel<<<dim3(D_DIM / 128, M / 128), 256, 0, stream>>>(
      h1, w2, M, D_DIM, F_DIM, 4, d_out, b2, yb);

  // 8) final norm in place on d_out
  zero_stats_kernel<<<1, 64, 0, stream>>>(st);
  reduce_stats_kernel<<<dim3(64, B_DIM), 256, 0, stream>>>((const float*)d_out, st);
  normalize_kernel<<<(B_DIM * DS) / 256, 256, 0, stream>>>((float*)d_out, st, nullptr);
}
